// KANLayer_15805479649621
// MI455X (gfx1250) — matrix operand order repaired and checked by reading, not compiled, not run on hardware
//
#include <hip/hip_runtime.h>
#include <hip/hip_bf16.h>

typedef __attribute__((ext_vector_type(16))) __bf16 bf16x16;
typedef __attribute__((ext_vector_type(8)))  __bf16 bf16x8;
typedef __attribute__((ext_vector_type(8)))  float  f32x8;

#define B_ROWS 16384
#define IN_F   1024
#define OUT_F  1024
#define GRID_G 3
#define KA     4096   // IN_F * (1 + GRID_G)
#define SE_HID 32

#define KTILE  64
#define LDA    72     // padded LDS row stride in bf16 elements (144B, 16B-aligned)

// ---------------------------------------------------------------------------
// Kernel 1: pack augmented weights  Waug[o][k], k<1024 -> base, else spline g
// ---------------------------------------------------------------------------
__global__ __launch_bounds__(256)
void pack_w_kernel(const float* __restrict__ base_w,
                   const float* __restrict__ spline_w,
                   __hip_bfloat16* __restrict__ Waug) {
    int idx = blockIdx.x * 256 + threadIdx.x;       // over OUT_F*KA = 4M
    if (idx >= OUT_F * KA) return;
    int o = idx / KA;
    int k = idx - o * KA;
    float v;
    if (k < IN_F) {
        v = base_w[o * IN_F + k];
    } else {
        int kk = k - IN_F;
        int g  = kk / IN_F;          // 0..2
        int i  = kk - g * IN_F;      // 0..1023
        v = spline_w[((size_t)o * IN_F + i) * GRID_G + g];
    }
    Waug[idx] = __float2bfloat16(v);
}

// ---------------------------------------------------------------------------
// Kernel 2: pack augmented activations Xaug[b][k] = [x | basis0 | basis1 | basis2]
// ---------------------------------------------------------------------------
__global__ __launch_bounds__(256)
void pack_x_kernel(const float* __restrict__ x,
                   __hip_bfloat16* __restrict__ Xaug) {
    int idx = blockIdx.x * 256 + threadIdx.x;       // over B_ROWS*IN_F = 16M
    if (idx >= B_ROWS * IN_F) return;
    int b = idx / IN_F;
    int i = idx - b * IN_F;
    float v = x[idx];
    float w[GRID_G];
#pragma unroll
    for (int g = 0; g < GRID_G; ++g) {
        float gv = -1.0f + (float)g;                // linspace(-1,1,3)
        float d  = fabsf(v - gv);
        float t  = 1.0f - d;
        w[g] = (d < 1.0f) ? t * t : 0.0f;
    }
    float inv = 1.0f / (w[0] + w[1] + w[2] + 1e-6f);
    size_t row = (size_t)b * KA;
    Xaug[row + i]               = __float2bfloat16(v);
    Xaug[row + IN_F + i]        = __float2bfloat16(w[0] * inv);
    Xaug[row + 2 * IN_F + i]    = __float2bfloat16(w[1] * inv);
    Xaug[row + 3 * IN_F + i]    = __float2bfloat16(w[2] * inv);
}

// ---------------------------------------------------------------------------
// Async copy helper: 64B (4 x b128) global -> LDS via CDNA5 async engine.
// the immediate offset is added to both the LDS and the global address, so a
// single base pair + immediate offsets covers all 4 chunks (no VALU adds).
// ---------------------------------------------------------------------------
__device__ __forceinline__ void async_cp64(uint32_t lds, const __bf16* g) {
#pragma unroll
    for (int c = 0; c < 4; ++c) {
        asm volatile("global_load_async_to_lds_b128 %0, %1, off offset:%2"
                     :: "v"(lds), "v"(g), "n"(c * 16) : "memory");
    }
}

// ---------------------------------------------------------------------------
// Kernel 3: Out[16384,1024] = Xaug[16384,4096] @ Waug[1024,4096]^T  (bf16 WMMA)
// Block: 256 thr = 8 waves (4 M x 2 N); block tile 128x128, K-tile 64.
// Async double-buffered LDS staging; fragments read from LDS.
// ---------------------------------------------------------------------------
__global__ __launch_bounds__(256)
void gemm_kernel(const __hip_bfloat16* __restrict__ Xaug_,
                 const __hip_bfloat16* __restrict__ Waug_,
                 float* __restrict__ Out) {
    __shared__ __align__(16) __bf16 sA[2][128 * LDA];
    __shared__ __align__(16) __bf16 sB[2][128 * LDA];

    const int tid   = threadIdx.x;
    const int lane  = tid & 31;
    const int wave  = tid >> 5;
    const int waveM = wave & 3;      // 0..3
    const int waveN = wave >> 2;     // 0..1
    const int l     = lane & 15;
    const int hv    = lane >> 4;

    const int Mblk = blockIdx.x * 128;
    const int Nblk = blockIdx.y * 128;

    const __bf16* X = reinterpret_cast<const __bf16*>(Xaug_);
    const __bf16* W = reinterpret_cast<const __bf16*>(Waug_);

    // Loader mapping: 2 threads per tile row; each copies half a row (32 elems).
    const int ldRow = tid >> 1;
    const int ldCol = (tid & 1) * 32;
    const __bf16* gA = X + (size_t)(Mblk + ldRow) * KA + ldCol;
    const __bf16* gB = W + (size_t)(Nblk + ldRow) * KA + ldCol;
    uint32_t lA[2], lB[2];
    lA[0] = (uint32_t)(size_t)&sA[0][ldRow * LDA + ldCol];
    lA[1] = (uint32_t)(size_t)&sA[1][ldRow * LDA + ldCol];
    lB[0] = (uint32_t)(size_t)&sB[0][ldRow * LDA + ldCol];
    lB[1] = (uint32_t)(size_t)&sB[1][ldRow * LDA + ldCol];

    f32x8 acc[2][4];
    const f32x8 zero = {0.f, 0.f, 0.f, 0.f, 0.f, 0.f, 0.f, 0.f};
#pragma unroll
    for (int mi = 0; mi < 2; ++mi)
#pragma unroll
        for (int ni = 0; ni < 4; ++ni) acc[mi][ni] = zero;

    // LDS fragment base offsets (elements)
    const int aOff0 = (waveM * 32 + l) * LDA + hv * 16;
    const int aOff1 = (waveM * 32 + 16 + l) * LDA + hv * 16;
    const int bOff  = (waveN * 64 + l) * LDA + hv * 16;

    constexpr int NSTEPS = KA / KTILE;   // 64

    // Prologue: two tiles in flight (8 async b128 per thread per tile)
    async_cp64(lA[0], gA);
    async_cp64(lB[0], gB);
    async_cp64(lA[1], gA + KTILE);
    async_cp64(lB[1], gB + KTILE);

    for (int ks = 0; ks < NSTEPS; ++ks) {
        const int buf = ks & 1;
        if (ks + 1 < NSTEPS)
            asm volatile("s_wait_asynccnt 0x8" ::: "memory");  // current tile done
        else
            asm volatile("s_wait_asynccnt 0x0" ::: "memory");  // last tile done
        __syncthreads();

        const __bf16* tA = &sA[buf][0];
        const __bf16* tB = &sB[buf][0];
#pragma unroll
        for (int kk = 0; kk < KTILE; kk += 32) {
            bf16x16 afr[2];
            {
                bf16x8 lo = *(const bf16x8*)(tA + aOff0 + kk);
                bf16x8 hi = *(const bf16x8*)(tA + aOff0 + kk + 8);
                afr[0] = __builtin_shufflevector(lo, hi,
                            0,1,2,3,4,5,6,7,8,9,10,11,12,13,14,15);
            }
            {
                bf16x8 lo = *(const bf16x8*)(tA + aOff1 + kk);
                bf16x8 hi = *(const bf16x8*)(tA + aOff1 + kk + 8);
                afr[1] = __builtin_shufflevector(lo, hi,
                            0,1,2,3,4,5,6,7,8,9,10,11,12,13,14,15);
            }
#pragma unroll
            for (int ni = 0; ni < 4; ++ni) {
                bf16x16 bfr = *(const bf16x16*)(tB + bOff + ni * 16 * LDA + kk);
#pragma unroll
                for (int mi = 0; mi < 2; ++mi) {
                    acc[mi][ni] = __builtin_amdgcn_wmma_f32_16x16x32_bf16(
                        false, afr[mi], false, bfr, (short)0, acc[mi][ni],
                        false, false);
                }
            }
        }
        __syncthreads();   // all waves done reading buf before overwrite

        if (ks + 2 < NSTEPS) {
            const __bf16* nA = gA + (size_t)(ks + 2) * KTILE;
            const __bf16* nB = gB + (size_t)(ks + 2) * KTILE;
            async_cp64(lA[buf], nA);
            async_cp64(lB[buf], nB);
        }
    }

    // C/D layout: lanes 0-15 -> N=lane, VGPR r = row M=r; lanes 16-31 -> M=8+r.
#pragma unroll
    for (int mi = 0; mi < 2; ++mi) {
        int mrow = Mblk + waveM * 32 + mi * 16 + hv * 8;
#pragma unroll
        for (int ni = 0; ni < 4; ++ni) {
            int ncol = Nblk + waveN * 64 + ni * 16 + l;
#pragma unroll
            for (int r = 0; r < 8; ++r) {
                Out[(size_t)(mrow + r) * OUT_F + ncol] = acc[mi][ni][r];
            }
        }
    }
}

// ---------------------------------------------------------------------------
// Kernel 4: per-row LayerNorm + SE gating, in place on Out. One block per row.
// ---------------------------------------------------------------------------
__global__ __launch_bounds__(256)
void epilogue_kernel(float* __restrict__ out,
                     const float* __restrict__ gamma,
                     const float* __restrict__ beta,
                     const float* __restrict__ w1,
                     const float* __restrict__ b1,
                     const float* __restrict__ w2,
                     const float* __restrict__ b2) {
    __shared__ float sN[OUT_F];
    __shared__ float sRed[16];
    __shared__ float sH[SE_HID];

    const int tid  = threadIdx.x;
    const int lane = tid & 31;
    const int wave = tid >> 5;
    float* orow = out + (size_t)blockIdx.x * OUT_F;

    float v[4], s = 0.f, ss = 0.f;
#pragma unroll
    for (int i = 0; i < 4; ++i) {
        v[i] = orow[tid + i * 256];
        s += v[i]; ss += v[i] * v[i];
    }
#pragma unroll
    for (int off = 16; off > 0; off >>= 1) {
        s  += __shfl_down(s,  off, 32);
        ss += __shfl_down(ss, off, 32);
    }
    if (lane == 0) { sRed[wave] = s; sRed[8 + wave] = ss; }
    __syncthreads();
    if (tid == 0) {
        float ts = 0.f, tss = 0.f;
        for (int w = 0; w < 8; ++w) { ts += sRed[w]; tss += sRed[8 + w]; }
        float mu  = ts / (float)OUT_F;
        float var = tss / (float)OUT_F - mu * mu;
        sRed[0] = mu;
        sRed[1] = rsqrtf(var + 1e-5f);
    }
    __syncthreads();
    float mu = sRed[0], rs = sRed[1];
    float nv[4];
#pragma unroll
    for (int i = 0; i < 4; ++i) {
        int o = tid + i * 256;
        nv[i] = (v[i] - mu) * rs * gamma[o] + beta[o];
        sN[o] = nv[i];
    }
    __syncthreads();

    // SE hidden: wave w computes hidden units j = 4w..4w+3 (1024-wide dots)
    {
        int jb = wave * 4;
        float p0 = 0.f, p1 = 0.f, p2 = 0.f, p3 = 0.f;
        for (int t = 0; t < 32; ++t) {
            int o = lane + t * 32;
            float nvv = sN[o];
            p0 += nvv * w1[(jb + 0) * OUT_F + o];
            p1 += nvv * w1[(jb + 1) * OUT_F + o];
            p2 += nvv * w1[(jb + 2) * OUT_F + o];
            p3 += nvv * w1[(jb + 3) * OUT_F + o];
        }
#pragma unroll
        for (int off = 16; off > 0; off >>= 1) {
            p0 += __shfl_down(p0, off, 32);
            p1 += __shfl_down(p1, off, 32);
            p2 += __shfl_down(p2, off, 32);
            p3 += __shfl_down(p3, off, 32);
        }
        if (lane == 0) {
            sH[jb + 0] = fmaxf(p0 + b1[jb + 0], 0.f);
            sH[jb + 1] = fmaxf(p1 + b1[jb + 1], 0.f);
            sH[jb + 2] = fmaxf(p2 + b1[jb + 2], 0.f);
            sH[jb + 3] = fmaxf(p3 + b1[jb + 3], 0.f);
        }
    }
    __syncthreads();

    // Gate: sigmoid(h @ w2^T + b2), elementwise multiply
#pragma unroll
    for (int i = 0; i < 4; ++i) {
        int o = tid + i * 256;
        float acc = b2[o];
        const float* wr = w2 + (size_t)o * SE_HID;
#pragma unroll
        for (int j = 0; j < SE_HID; ++j) acc += sH[j] * wr[j];
        float gate = 1.0f / (1.0f + __expf(-acc));
        orow[o] = nv[i] * gate;
    }
}

// ---------------------------------------------------------------------------
extern "C" void kernel_launch(void* const* d_in, const int* in_sizes, int n_in,
                              void* d_out, int out_size, void* d_ws, size_t ws_size,
                              hipStream_t stream) {
    const float* x     = (const float*)d_in[0];
    const float* bw    = (const float*)d_in[1];
    const float* sw    = (const float*)d_in[2];
    const float* gamma = (const float*)d_in[3];
    const float* beta  = (const float*)d_in[4];
    const float* w1    = (const float*)d_in[5];
    const float* b1    = (const float*)d_in[6];
    const float* w2    = (const float*)d_in[7];
    const float* b2    = (const float*)d_in[8];
    float* out = (float*)d_out;

    __hip_bfloat16* Xaug = (__hip_bfloat16*)d_ws;
    __hip_bfloat16* Waug =
        (__hip_bfloat16*)((char*)d_ws + (size_t)B_ROWS * KA * sizeof(__hip_bfloat16));

    pack_w_kernel<<<(OUT_F * KA) / 256, 256, 0, stream>>>(bw, sw, Waug);
    pack_x_kernel<<<(B_ROWS * IN_F) / 256, 256, 0, stream>>>(x, Xaug);
    gemm_kernel<<<dim3(B_ROWS / 128, OUT_F / 128), 256, 0, stream>>>(Xaug, Waug, out);
    epilogue_kernel<<<B_ROWS, 256, 0, stream>>>(out, gamma, beta, w1, b1, w2, b2);
}
